// EntNet_5497558139413
// MI455X (gfx1250) — matrix operand order repaired and checked by reading, not compiled, not run on hardware
//
#include <hip/hip_runtime.h>
#include <hip/hip_bf16.h>
#include <math.h>

#define Tn  512
#define Bn  128
#define En  128
#define Jn  20
#define Ln  3
#define E3n 384

typedef __attribute__((ext_vector_type(16))) __bf16          v16bf;
typedef __attribute__((ext_vector_type(16))) unsigned short  v16us;
typedef __attribute__((ext_vector_type(8)))  float           v8f;

__device__ __forceinline__ unsigned short f2bf(float f){
  unsigned int u = __float_as_uint(f);
  u += 0x7FFFu + ((u >> 16) & 1u);          // round-to-nearest-even
  return (unsigned short)(u >> 16);
}
__device__ __forceinline__ unsigned int pk2(float lo, float hi){
  return (unsigned int)f2bf(lo) | ((unsigned int)f2bf(hi) << 16);
}
__device__ __forceinline__ float sigmoidf_(float x){
  return 1.0f / (1.0f + __expf(-x));
}
__device__ __forceinline__ v8f wmma_bf16(v16bf a, v16bf b, v8f c){
  return __builtin_amdgcn_wmma_f32_16x16x32_bf16(false, a, false, b, (short)0, c, false, false);
}
// A fragment (16x32 slice of a 16x128 bf16 LDS tile) -- pure ds loads.
// lanes0-15: K = kc*32 + {0..7, 16..23}; lanes16-31: +8.
__device__ __forceinline__ v16bf fragA_bf(const unsigned short* Ab, int kc, int mrow, int kbA){
  const unsigned short* r = Ab + mrow*En + kc*32 + kbA;
  v16us h;
#pragma unroll
  for (int i=0;i<8;i++) h[i]   = r[i];
#pragma unroll
  for (int i=0;i<8;i++) h[8+i] = r[16+i];
  return __builtin_bit_cast(v16bf, h);
}
// B fragment (32x16): weights stored transposed [n][k] bf16 in LDS.
// lanes0-15 -> K=kc*32+{0..7, 16..23}, lanes16-31 -> +8.
__device__ __forceinline__ v16bf fragB_bf(const unsigned short* BT, int col, int kc, int kbB){
  const unsigned short* r = BT + col*En + kc*32 + kbB;
  v16us h;
#pragma unroll
  for (int i=0;i<8;i++) h[i] = r[i];
#pragma unroll
  for (int i=0;i<8;i++) h[8+i] = r[16+i];
  return __builtin_bit_cast(v16bf, h);
}

// ---------------------------------------------------------------------------
// Pack: bf16-cast (transpose where needed) weights; kV = keys@Vm; kWa = keys@W_att
// ---------------------------------------------------------------------------
__global__ void entnet_pack(
    const float* __restrict__ U, const float* __restrict__ W,
    const float* __restrict__ Wih, const float* __restrict__ Whh,
    const float* __restrict__ keys, const float* __restrict__ Vm,
    const float* __restrict__ W_att,
    unsigned short* __restrict__ Ubf, unsigned short* __restrict__ Wbf,
    unsigned short* __restrict__ Wihbf, unsigned short* __restrict__ Whhbf,
    float* __restrict__ kV, float* __restrict__ kWa)
{
  int id = blockIdx.x*blockDim.x + threadIdx.x;
  if (id < 16384){ int f = id >> 7, e = id & 127; Ubf[id] = f2bf(U[e*En + f]); return; }
  id -= 16384;
  if (id < 16384){ int f = id >> 7, e = id & 127; Wbf[id] = f2bf(W[e*En + f]); return; }
  id -= 16384;
  if (id < 49152){ Wihbf[id] = f2bf(Wih[id]); return; }   // x@Wih^T -> B^T layout == Wih
  id -= 49152;
  if (id < 49152){ Whhbf[id] = f2bf(Whh[id]); return; }
  id -= 49152;
  if (id < 2560){
    int jj = id >> 7, e = id & 127;
    float s = 0.f;
    for (int k=0;k<En;k++) s += keys[jj*En+k]*Vm[k*En+e];
    kV[id] = s; return;
  }
  id -= 2560;
  if (id < 5120){
    int jj = id >> 8, e = id & 255;
    float s = 0.f;
    for (int k=0;k<En;k++) s += keys[jj*En+k]*W_att[k*2*En+e];
    kWa[id] = s;
  }
}

// ---------------------------------------------------------------------------
// Main persistent scan: 1 WG (256 thr, 8 waves) = one (j, 16-row) tile, 512 steps.
// ---------------------------------------------------------------------------
__global__ __launch_bounds__(256) void entnet_main(
    const int* __restrict__ text, const float* __restrict__ emb,
    const float* __restrict__ keys, const float* __restrict__ U_bias,
    const float* __restrict__ vv, const float* __restrict__ bih,
    const float* __restrict__ bhh, const float* __restrict__ h0,
    const float* __restrict__ d0,
    const unsigned short* __restrict__ gU, const unsigned short* __restrict__ gW,
    const unsigned short* __restrict__ gWih, const unsigned short* __restrict__ gWhh,
    const float* __restrict__ gkV, float* __restrict__ hfin)
{
  __shared__ __align__(16) unsigned short sU[En*En];       // U^T  bf16 [n][k]
  __shared__ __align__(16) unsigned short sW[En*En];       // W^T  bf16 [n][k]
  __shared__ __align__(16) unsigned short sWih[E3n*En];    // Wih  bf16 [f][e]
  __shared__ __align__(16) unsigned short sWhh[E3n*En];
  __shared__ __align__(16) float sh[16*En];                // h fp32 (in-place)
  __shared__ __align__(16) float sd[16*En];                // d fp32 (in-place)
  __shared__ __align__(16) float scandi[16*En];
  __shared__ __align__(16) float swrow[16*En];
  __shared__ __align__(16) unsigned short sh_bf[16*En];    // bf16 shadows (GEMM A inputs)
  __shared__ __align__(16) unsigned short sd_bf[16*En];
  __shared__ __align__(16) unsigned short scandi_bf[16*En];
  __shared__ __align__(16) unsigned short swrow_bf[16*En];
  __shared__ float scb[En], skey[En], sv[En];
  __shared__ float sbih[E3n], sbhh[E3n];
  __shared__ float sred[256];
  __shared__ float sapb[16], sgate[16], srnorm[16];

  const int tid  = threadIdx.x;
  const int lane = tid & 31;
  const int wid  = tid >> 5;          // 0..7
  const int mrow = lane & 15;
  const int mb   = (lane < 16) ? 0 : 8;
  const int kbA  = (lane < 16) ? 0 : 8;
  const int kbB  = (lane < 16) ? 0 : 8;
  const int j    = blockIdx.x >> 3;
  const int b0   = (blockIdx.x & 7) << 4;
  const int er   = tid >> 4;          // elementwise row 0..15
  const int ec   = (tid & 15) << 3;   // elementwise col base (8 elems each)

  { // resident weights -> LDS (256KB bf16, once)
    const uint4* g; uint4* s;
    g=(const uint4*)gU;   s=(uint4*)sU;   for (int i=tid;i<2048;i+=256) s[i]=g[i];
    g=(const uint4*)gW;   s=(uint4*)sW;   for (int i=tid;i<2048;i+=256) s[i]=g[i];
    g=(const uint4*)gWih; s=(uint4*)sWih; for (int i=tid;i<6144;i+=256) s[i]=g[i];
    g=(const uint4*)gWhh; s=(uint4*)sWhh; for (int i=tid;i<6144;i+=256) s[i]=g[i];
  }
  if (tid < En){
    scb[tid]  = U_bias[tid] + gkV[j*En + tid];   // U_bias + kV[j]
    skey[tid] = keys[j*En + tid];
    sv[tid]   = vv[tid];
  }
  for (int i=tid;i<E3n;i+=256){ sbih[i]=bih[i]; sbhh[i]=bhh[i]; }
  for (int i=tid;i<2048;i+=256){
    int r=i>>7, e=i&127;
    size_t gi = ((size_t)j*Bn + b0 + r)*En + e;
    float hv = h0[gi], dv = d0[gi];
    sh[i]=hv; sd[i]=dv;
    sh_bf[i]=f2bf(hv); sd_bf[i]=f2bf(dv);
  }
  __syncthreads();

  for (int t=0; t<Tn; ++t){
    // phase 1: gather 16 embedding rows (fp32 + bf16 shadow); prefetch next step
    {
      size_t idx = (size_t)text[t*Bn + b0 + er];
      const float4* src = (const float4*)(emb + idx*En + ec);
      float4 f0 = src[0], f1 = src[1];
      float4* dst = (float4*)(swrow + er*En + ec);
      dst[0]=f0; dst[1]=f1;
      uint4 q;
      q.x = pk2(f0.x,f0.y); q.y = pk2(f0.z,f0.w);
      q.z = pk2(f1.x,f1.y); q.w = pk2(f1.z,f1.w);
      *(uint4*)(swrow_bf + er*En + ec) = q;
      if (t+1 < Tn){
        size_t idx2 = (size_t)text[(t+1)*Bn + b0 + er];
        __builtin_prefetch(emb + idx2*En + ec, 0, 1);   // global_prefetch
      }
    }
    __syncthreads();

    // phase 2: a[r] + bgate[r] = w.h + keys_j.w  (partials then 16-row reduce)
    {
      const float* wr = swrow + er*En + ec;
      const float* hr = sh    + er*En + ec;
      float s = 0.f;
#pragma unroll
      for (int i=0;i<8;i++) s += wr[i]*(hr[i] + skey[ec+i]);
      sred[tid] = s;
    }
    __syncthreads();
    if (tid < 16){
      float s=0.f;
#pragma unroll
      for (int k=0;k<16;k++) s += sred[tid*16+k];
      sapb[tid] = s;
    }

    // old-d A fragments: load BEFORE d gets overwritten (stable since last step)
    v16bf da[4];
#pragma unroll
    for (int kc=0;kc<4;kc++) da[kc] = fragA_bf(sd_bf, kc, mrow, kbA);

    // phase 3: candi = relu(h@U + w@W + (U_bias + kV_j)), one fused accumulator
    {
      v16bf ha[4], wa[4];
#pragma unroll
      for (int kc=0;kc<4;kc++){
        ha[kc] = fragA_bf(sh_bf,    kc, mrow, kbA);
        wa[kc] = fragA_bf(swrow_bf, kc, mrow, kbA);
      }
      int coln = wid*16 + mrow;               // this lane's output column
      v8f acc = {};
#pragma unroll
      for (int kc=0;kc<4;kc++) acc = wmma_bf16(ha[kc], fragB_bf(sU, coln, kc, kbB), acc);
#pragma unroll
      for (int kc=0;kc<4;kc++) acc = wmma_bf16(wa[kc], fragB_bf(sW, coln, kc, kbB), acc);
      float cb = scb[coln];
#pragma unroll
      for (int r=0;r<8;r++){
        float val = fmaxf(acc[r] + cb, 0.f);
        scandi[(mb+r)*En + coln]    = val;
        scandi_bf[(mb+r)*En + coln] = f2bf(val);
      }
    }
    __syncthreads();

    // phase 4: gi = candi@Wih^T, gh = d@Whh^T, GRU elementwise in regs (in-place d)
    {
      v16bf ca[4];
#pragma unroll
      for (int kc=0;kc<4;kc++) ca[kc] = fragA_bf(scandi_bf, kc, mrow, kbA);
      int fn = wid*16 + mrow;                 // this lane's f column (0..127)
      v8f gi0={},gi1={},gi2={},gh0={},gh1={},gh2={};
#pragma unroll
      for (int kc=0;kc<4;kc++){
        gi0 = wmma_bf16(ca[kc], fragB_bf(sWih, fn,        kc, kbB), gi0);
        gi1 = wmma_bf16(ca[kc], fragB_bf(sWih, En + fn,   kc, kbB), gi1);
        gi2 = wmma_bf16(ca[kc], fragB_bf(sWih, 2*En + fn, kc, kbB), gi2);
        gh0 = wmma_bf16(da[kc], fragB_bf(sWhh, fn,        kc, kbB), gh0);
        gh1 = wmma_bf16(da[kc], fragB_bf(sWhh, En + fn,   kc, kbB), gh1);
        gh2 = wmma_bf16(da[kc], fragB_bf(sWhh, 2*En + fn, kc, kbB), gh2);
      }
      float bi0=sbih[fn], bi1=sbih[En+fn], bi2=sbih[2*En+fn];
      float bh0=sbhh[fn], bh1=sbhh[En+fn], bh2=sbhh[2*En+fn];
#pragma unroll
      for (int r=0;r<8;r++){
        float ir = gi0[r]+bi0, iz = gi1[r]+bi1, in_ = gi2[r]+bi2;
        float hr = gh0[r]+bh0, hz = gh1[r]+bh1, hn  = gh2[r]+bh2;
        float rg = sigmoidf_(ir+hr);
        float zg = sigmoidf_(iz+hz);
        float ng = tanhf(in_ + rg*hn);
        int m = mb + r;
        float dold = sd[m*En + fn];          // each (m,fn) owned by exactly this lane
        float dn = (1.f - zg)*ng + zg*dold;
        sd[m*En + fn]    = dn;
        sd_bf[m*En + fn] = f2bf(dn);
      }
    }
    __syncthreads();

    // phase 5: c[r] = new_d . v ; gate = sigmoid(a + bgate + c)
    {
      const float* dr = sd + er*En + ec;
      float s = 0.f;
#pragma unroll
      for (int i=0;i<8;i++) s += dr[i]*sv[ec+i];
      sred[tid] = s;
    }
    __syncthreads();
    if (tid < 16){
      float s=0.f;
#pragma unroll
      for (int k=0;k<16;k++) s += sred[tid*16+k];
      sgate[tid] = sigmoidf_(sapb[tid] + s);
    }
    __syncthreads();

    // phase 6: h = normalize(h + gate*candi)  (in-place, per-element ownership)
    {
      float g = sgate[er];
      float s = 0.f;
#pragma unroll
      for (int i=0;i<8;i++){
        float hv = sh[er*En + ec + i] + g*scandi[er*En + ec + i];
        sh[er*En + ec + i] = hv;
        s += hv*hv;
      }
      sred[tid] = s;
    }
    __syncthreads();
    if (tid < 16){
      float s=0.f;
#pragma unroll
      for (int k=0;k<16;k++) s += sred[tid*16+k];
      srnorm[tid] = rsqrtf(s);
    }
    __syncthreads();
    {
      float rn = srnorm[er];
      float w0 = sh[er*En+ec+0]*rn, w1 = sh[er*En+ec+1]*rn;
      float w2 = sh[er*En+ec+2]*rn, w3 = sh[er*En+ec+3]*rn;
      float w4 = sh[er*En+ec+4]*rn, w5 = sh[er*En+ec+5]*rn;
      float w6 = sh[er*En+ec+6]*rn, w7 = sh[er*En+ec+7]*rn;
      sh[er*En+ec+0]=w0; sh[er*En+ec+1]=w1; sh[er*En+ec+2]=w2; sh[er*En+ec+3]=w3;
      sh[er*En+ec+4]=w4; sh[er*En+ec+5]=w5; sh[er*En+ec+6]=w6; sh[er*En+ec+7]=w7;
      uint4 q;
      q.x = pk2(w0,w1); q.y = pk2(w2,w3); q.z = pk2(w4,w5); q.w = pk2(w6,w7);
      *(uint4*)(sh_bf + er*En + ec) = q;
    }
    __syncthreads();
  }

  for (int i=tid;i<2048;i+=256){
    int r=i>>7, e=i&127;
    hfin[((size_t)j*Bn + b0 + r)*En + e] = sh[i];
  }
}

// ---------------------------------------------------------------------------
// Attention + u + c1 (one block per batch element)
// ---------------------------------------------------------------------------
__global__ __launch_bounds__(128) void entnet_att(
    const int* __restrict__ target, const int* __restrict__ aspect,
    const float* __restrict__ emb, const float* __restrict__ kWa,
    const float* __restrict__ hfin, const float* __restrict__ c1_w,
    const float* __restrict__ c1_b, float* __restrict__ hidden)
{
  __shared__ float sta[2*En];
  __shared__ float satt[Jn];
  __shared__ float su[En];
  int b = blockIdx.x, tid = threadIdx.x;
  sta[tid]      = emb[(size_t)target[b]*En + tid];
  sta[En + tid] = emb[(size_t)aspect[b]*En + tid];
  __syncthreads();
  if (tid < Jn){
    float s = 0.f;
    for (int k=0;k<2*En;k++) s += kWa[tid*2*En + k]*sta[k];
    satt[tid] = s;
  }
  __syncthreads();
  if (tid == 0){
    float m = satt[0];
    for (int k=1;k<Jn;k++) m = fmaxf(m, satt[k]);
    float s = 0.f;
    for (int k=0;k<Jn;k++){ satt[k] = __expf(satt[k]-m); s += satt[k]; }
    float inv = 1.f/s;
    for (int k=0;k<Jn;k++) satt[k] *= inv;
  }
  __syncthreads();
  float u = 0.f;
  for (int jj=0;jj<Jn;jj++) u += satt[jj]*hfin[((size_t)jj*Bn + b)*En + tid];
  su[tid] = u;
  __syncthreads();
  float hd = c1_b[tid] + sta[En + tid];        // + aspect_embed
  for (int k=0;k<En;k++) hd += su[k]*c1_w[tid*En + k];
  hidden[b*En + tid] = hd;
}

// ---------------------------------------------------------------------------
// Batchnorm (over batch) + relu + c2 (single block)
// ---------------------------------------------------------------------------
__global__ __launch_bounds__(128) void entnet_bnout(
    const float* __restrict__ hidden, const float* __restrict__ bn_g,
    const float* __restrict__ bn_b, const float* __restrict__ c2_w,
    const float* __restrict__ c2_b, float* __restrict__ out)
{
  __shared__ float shid[Bn*En];
  int tid = threadIdx.x;
  {
    int e = tid;
    float s1=0.f, s2=0.f;
    for (int b=0;b<Bn;b++){ float x = hidden[b*En+e]; s1 += x; s2 += x*x; }
    float mu  = s1 / (float)Bn;
    float var = s2 / (float)Bn - mu*mu;
    float sc  = rsqrtf(var + 1e-5f)*bn_g[e];
    float bb  = bn_b[e];
    for (int b=0;b<Bn;b++){
      float x = (hidden[b*En+e]-mu)*sc + bb;
      shid[b*En+e] = fmaxf(x, 0.f);
    }
  }
  __syncthreads();
  {
    int b = tid;
    for (int l=0;l<Ln;l++){
      float s = c2_b[l];
      for (int e=0;e<En;e++) s += shid[b*En+e]*c2_w[l*En+e];
      out[b*Ln + l] = s;
    }
  }
}

extern "C" void kernel_launch(void* const* d_in, const int* in_sizes, int n_in,
                              void* d_out, int out_size, void* d_ws, size_t ws_size,
                              hipStream_t stream)
{
  (void)in_sizes; (void)n_in; (void)out_size; (void)ws_size;
  const int*   text   = (const int*)  d_in[0];
  const int*   target = (const int*)  d_in[1];
  const int*   aspect = (const int*)  d_in[2];
  const float* emb    = (const float*)d_in[3];
  const float* keys   = (const float*)d_in[4];
  const float* U      = (const float*)d_in[5];
  const float* Vm     = (const float*)d_in[6];
  const float* W      = (const float*)d_in[7];
  const float* U_bias = (const float*)d_in[8];
  const float* v      = (const float*)d_in[9];
  const float* Wih    = (const float*)d_in[10];
  const float* Whh    = (const float*)d_in[11];
  const float* bih    = (const float*)d_in[12];
  const float* bhh    = (const float*)d_in[13];
  const float* W_att  = (const float*)d_in[14];
  const float* c1_w   = (const float*)d_in[15];
  const float* c1_b   = (const float*)d_in[16];
  const float* bn_g   = (const float*)d_in[17];
  const float* bn_b   = (const float*)d_in[18];
  const float* c2_w   = (const float*)d_in[19];
  const float* c2_b   = (const float*)d_in[20];
  const float* h0     = (const float*)d_in[21];
  const float* d0     = (const float*)d_in[22];

  char* ws = (char*)d_ws;
  size_t o = 0;
  unsigned short* wUbf   = (unsigned short*)(ws + o); o += 16384*sizeof(unsigned short);
  unsigned short* wWbf   = (unsigned short*)(ws + o); o += 16384*sizeof(unsigned short);
  unsigned short* wWihbf = (unsigned short*)(ws + o); o += 49152*sizeof(unsigned short);
  unsigned short* wWhhbf = (unsigned short*)(ws + o); o += 49152*sizeof(unsigned short);
  float* wkV   = (float*)(ws + o); o += 2560*sizeof(float);
  float* wkWa  = (float*)(ws + o); o += 5120*sizeof(float);
  float* whfin = (float*)(ws + o); o += (size_t)Jn*Bn*En*sizeof(float);
  float* whid  = (float*)(ws + o); o += (size_t)Bn*En*sizeof(float);

  entnet_pack<<<dim3(542), dim3(256), 0, stream>>>(U, W, Wih, Whh, keys, Vm, W_att,
      wUbf, wWbf, wWihbf, wWhhbf, wkV, wkWa);
  entnet_main<<<dim3(Jn*(Bn/16)), dim3(256), 0, stream>>>(text, emb, keys, U_bias, v,
      bih, bhh, h0, d0, wUbf, wWbf, wWihbf, wWhhbf, wkV, whfin);
  entnet_att<<<dim3(Bn), dim3(128), 0, stream>>>(target, aspect, emb, wkWa, whfin,
      c1_w, c1_b, whid);
  entnet_bnout<<<dim3(1), dim3(128), 0, stream>>>(whid, bn_g, bn_b, c2_w, c2_b,
      (float*)d_out);
}
